// TimesBlock_31473520345364
// MI455X (gfx1250) — compile-verified, matrix operand order checked by reading, not run on hardware
//
#include <hip/hip_runtime.h>

#define TB_B    8
#define TB_L    4096
#define TB_C    128
#define TB_HID  256
#define TB_NK   3
#define TB_M    (TB_B * TB_L)     // 32768 rows total
#define TB_LMAX 6144              // max Np*p  (< L + p, p <= 2048)

typedef __attribute__((ext_vector_type(16))) _Float16 v16h;
typedef __attribute__((ext_vector_type(8)))  _Float16 v8h;
typedef __attribute__((ext_vector_type(8)))  float    v8f;

// pointer types for the gfx1250 async global->LDS builtin:
// param 0: global (as1) pointer to 16-byte int vector; param 1: LDS (as3) same.
typedef __attribute__((__vector_size__(4 * sizeof(int)))) int tb_v4i;
typedef __attribute__((address_space(1))) tb_v4i* tb_gp128;
typedef __attribute__((address_space(3))) tb_v4i* tb_lp128;

static __device__ __forceinline__ float tb_gelu(float x) {
  return 0.5f * x * (1.0f + erff(x * 0.7071067811865476f));
}
static __device__ __forceinline__ float tb_sigmoid(float x) {
  return 1.0f / (1.0f + expf(-x));
}

// ---- WMMA 16x16x32 f16 fragment loader (wave32 layout per CDNA5 ISA) ------
// For both A (row-major MxK, lane holds row m=lane&15) and B in N-major
// (N x K "weight [out,in]" layout, lane holds row n=lane&15):
// k-base = 8*(lane>>4); half i -> k = 16*(i/8) + kb + (i%8)
// => per lane: two contiguous 16-byte chunks -> 2x b128 loads.
static __device__ __forceinline__ v16h tb_frag(const _Float16* p, int ld, int lane) {
  const _Float16* row = p + (size_t)(lane & 15) * ld + (((lane >> 4) << 3));
  const v8h lo = *(const v8h*)(row);
  const v8h hi = *(const v8h*)(row + 16);
  v16h r;
#pragma unroll
  for (int i = 0; i < 8; ++i) { r[i] = lo[i]; r[i + 8] = hi[i]; }
  return r;
}

#define TB_WMMA(a, b, c) \
  __builtin_amdgcn_wmma_f32_16x16x32_f16(false, (a), false, (b), (short)0, (c), false, false)

// ---- cooperative global->LDS staging of a block-shared tile ----------------
// Uses gfx1250 async global->LDS (ASYNCcnt) when available; vector-copy else.
static __device__ __forceinline__ void tb_stage(_Float16* dst, const _Float16* src, int nhalf) {
  const int nvec = nhalf >> 3;
#if __has_builtin(__builtin_amdgcn_global_load_async_to_lds_b128)
  for (int i = threadIdx.x; i < nvec; i += 256) {
    __builtin_amdgcn_global_load_async_to_lds_b128(
        (tb_gp128)(uintptr_t)((const v8h*)src + i),
        (tb_lp128)(uintptr_t)((v8h*)dst + i),
        0, 0);
  }
#if __has_builtin(__builtin_amdgcn_s_wait_asynccnt)
  __builtin_amdgcn_s_wait_asynccnt(0);
#else
  asm volatile("s_wait_asynccnt 0x0" ::: "memory");
#endif
  __syncthreads();
#else
  for (int i = threadIdx.x; i < nvec; i += 256) {
    ((v8h*)dst)[i] = ((const v8h*)src)[i];
  }
  __syncthreads();
#endif
}

// ---------------- 0) weight prep: f32 -> f16, N-major (out,in) --------------
// transpose KxN (f32) -> NxK (f16)
__global__ __launch_bounds__(256) void tb_prep_t(const float* __restrict__ w,
                                                 _Float16* __restrict__ wh,
                                                 int K, int N) {
  const int i = blockIdx.x * 256 + threadIdx.x;   // over N*K in NK order
  const int n = i / K, k = i - n * K;
  wh[i] = (_Float16)w[(size_t)k * N + n];
}
// straight cast (already N-major [out,in])
__global__ __launch_bounds__(256) void tb_prep_c(const float* __restrict__ w,
                                                 _Float16* __restrict__ wh) {
  const int i = blockIdx.x * 256 + threadIdx.x;
  wh[i] = (_Float16)w[i];
}

// ---------------- 1) RMS norm: x (f32) -> xn (f16) --------------------------
__global__ __launch_bounds__(256) void tb_rms(const float* __restrict__ x,
                                              const float* __restrict__ w,
                                              _Float16* __restrict__ xn) {
  const int row  = blockIdx.x * 8 + (threadIdx.x >> 5);
  const int lane = threadIdx.x & 31;
  const float* xr = x + (size_t)row * TB_C;
  float s = 0.f;
#pragma unroll
  for (int i = 0; i < TB_C / 32; ++i) { const float v = xr[lane + 32 * i]; s += v * v; }
#pragma unroll
  for (int off = 16; off > 0; off >>= 1) s += __shfl_down(s, off, 32);
  s = __shfl(s, 0, 32);
  const float inv = rsqrtf(s * (1.0f / TB_C) + 1e-5f);
#pragma unroll
  for (int i = 0; i < TB_C / 32; ++i) {
    const int c = lane + 32 * i;
    xn[(size_t)row * TB_C + c] = (_Float16)(xr[c] * inv * w[c]);
  }
}

// ---------------- 2) GLU GEMM: xn (Mx128) @ glu_W (128x256) + sigmoid gate ---------
__global__ __launch_bounds__(256) void tb_glu(const _Float16* __restrict__ xn,
                                              const _Float16* __restrict__ gwh, // (256,128) NK
                                              const float* __restrict__ gB,
                                              _Float16* __restrict__ h1) {
  __shared__ __attribute__((aligned(16))) _Float16 sA[16 * TB_C];
  const int mt = blockIdx.x;
  const int wv = threadIdx.x >> 5;
  const int lane = threadIdx.x & 31;
  tb_stage(sA, xn + (size_t)mt * 16 * TB_C, 16 * TB_C);
  v8f ca = {}, cg = {};
#pragma unroll
  for (int ks = 0; ks < 4; ++ks) {
    v16h a  = tb_frag(sA + ks * 32, TB_C, lane);
    v16h b0 = tb_frag(gwh + (size_t)(wv * 16) * TB_C + ks * 32, TB_C, lane);
    v16h b1 = tb_frag(gwh + (size_t)(TB_C + wv * 16) * TB_C + ks * 32, TB_C, lane);
    ca = TB_WMMA(a, b0, ca);
    cg = TB_WMMA(a, b1, cg);
  }
  const int n  = lane & 15;
  const int rb = (lane >> 4) << 3;
  const float ba = gB[wv * 16 + n];
  const float bg = gB[TB_C + wv * 16 + n];
#pragma unroll
  for (int r = 0; r < 8; ++r) {
    const int row = mt * 16 + rb + r;
    const float av = ca[r] + ba;
    const float gv = cg[r] + bg;
    h1[(size_t)row * TB_C + wv * 16 + n] = (_Float16)(av * tb_sigmoid(gv));
  }
}

// ---------------- 3) pin GEMM: h1 (Mx128) @ pin_W (128x128) -> hf (f32) ----
__global__ __launch_bounds__(256) void tb_pin(const _Float16* __restrict__ h1,
                                              const _Float16* __restrict__ pWh, // (128,128) NK
                                              const float* __restrict__ pB,
                                              float* __restrict__ hf) {
  __shared__ __attribute__((aligned(16))) _Float16 sA[16 * TB_C];
  const int mt = blockIdx.x;
  const int wv = threadIdx.x >> 5;
  const int lane = threadIdx.x & 31;
  tb_stage(sA, h1 + (size_t)mt * 16 * TB_C, 16 * TB_C);
  v8f c = {};
#pragma unroll
  for (int ks = 0; ks < 4; ++ks) {
    v16h a = tb_frag(sA + ks * 32, TB_C, lane);
    v16h b = tb_frag(pWh + (size_t)(wv * 16) * TB_C + ks * 32, TB_C, lane);
    c = TB_WMMA(a, b, c);
  }
  const int n  = lane & 15;
  const int rb = (lane >> 4) << 3;
  const float bias = pB[wv * 16 + n];
#pragma unroll
  for (int r = 0; r < 8; ++r) {
    const int row = mt * 16 + rb + r;
    hf[(size_t)row * TB_C + wv * 16 + n] = c[r] + bias;
  }
}

// ---------------- 4) per-(b,c) mean over L ---------------------------------
__global__ __launch_bounds__(256) void tb_mean(const float* __restrict__ hf,
                                               float* __restrict__ mean) {
  const int idx  = blockIdx.x * 8 + (threadIdx.x >> 5);
  const int lane = threadIdx.x & 31;
  const int b = idx >> 7, c = idx & 127;
  float s = 0.f;
  for (int l = lane; l < TB_L; l += 32) s += hf[((size_t)b * TB_L + l) * TB_C + c];
#pragma unroll
  for (int off = 16; off > 0; off >>= 1) s += __shfl_down(s, off, 32);
  if (lane == 0) mean[b * TB_C + c] = s * (1.0f / TB_L);
}

// ---------------- 5) x0 = hf - mean (f16); zero agg ------------------------
__global__ __launch_bounds__(256) void tb_x0agg(const float* __restrict__ hf,
                                                const float* __restrict__ mean,
                                                _Float16* __restrict__ x0,
                                                float* __restrict__ agg) {
  const size_t i = (size_t)blockIdx.x * 256 + threadIdx.x;
  const int c = (int)(i & (TB_C - 1));
  const int b = (int)(i >> 19);   // L*C = 2^19
  x0[i]  = (_Float16)(hf[i] - mean[b * TB_C + c]);
  agg[i] = 0.f;
}

__global__ __launch_bounds__(256) void tb_zero_ac(float* __restrict__ ac) {
  ac[blockIdx.x * 256 + threadIdx.x] = 0.f;
}

// ---------------- 6) circular autocorrelation via WMMA Gram tiles ----------
// ac[t] = sum_l <x0[l], x0[(l+t) mod L]>  ==  sum over G[m,n] with (n-m)%L == t
__global__ __launch_bounds__(256) void tb_autocorr(const _Float16* __restrict__ x0,
                                                   float* __restrict__ ac) {
  __shared__ float lac[TB_L];                       // 16 KB
  const int b  = blockIdx.x >> 8;
  const int it = blockIdx.x & 255;                  // I tile (rows m)
  const int wv = threadIdx.x >> 5;
  const int lane = threadIdx.x & 31;
  for (int i = threadIdx.x; i < TB_L; i += 256) lac[i] = 0.f;
  __syncthreads();
  const _Float16* xb = x0 + (size_t)b * TB_L * TB_C;
  v16h afr[4];
#pragma unroll
  for (int ks = 0; ks < 4; ++ks)
    afr[ks] = tb_frag(xb + (size_t)it * 16 * TB_C + ks * 32, TB_C, lane);
  const int n  = lane & 15;
  const int rb = (lane >> 4) << 3;
  for (int jt = wv; jt < 256; jt += 8) {            // J tile (cols n'), wave-uniform
    if (jt + 8 < 256)
      __builtin_prefetch((const void*)(xb + (size_t)(jt + 8) * 16 * TB_C), 0, 1);
    v8f g = {};
#pragma unroll
    for (int ks = 0; ks < 4; ++ks) {
      v16h bf = tb_frag(xb + (size_t)jt * 16 * TB_C + ks * 32, TB_C, lane);
      g = TB_WMMA(afr[ks], bf, g);
    }
#pragma unroll
    for (int r = 0; r < 8; ++r) {
      const int mm  = it * 16 + rb + r;
      const int nn  = jt * 16 + n;
      const int lag = (nn - mm) & (TB_L - 1);
      if (lag >= 2 && lag <= 2048) atomicAdd(&lac[lag], g[r]);
    }
  }
  __syncthreads();
  for (int i = threadIdx.x; i < TB_L; i += 256) {
    const float v = lac[i];
    if (v != 0.f) atomicAdd(&ac[(size_t)b * TB_L + i], v);
  }
}

// ---------------- 7) top-3 lags per batch ----------------------------------
__global__ __launch_bounds__(256) void tb_topk(const float* __restrict__ ac,
                                               int* __restrict__ per) {
  const int b   = blockIdx.x;
  const int tid = threadIdx.x;
  __shared__ float sv[256];
  __shared__ int   si[256];
  __shared__ int   chosen[TB_NK];
  const float* a = ac + (size_t)b * TB_L;
  for (int pass = 0; pass < TB_NK; ++pass) {
    float best = -3.4e38f; int bi = 2;
    for (int lag = 2 + tid; lag <= 2048; lag += 256) {
      bool skip = false;
      for (int j = 0; j < pass; ++j) if (chosen[j] == lag) skip = true;
      const float v = a[lag];
      if (!skip && v > best) { best = v; bi = lag; }
    }
    sv[tid] = best; si[tid] = bi;
    __syncthreads();
    for (int s = 128; s > 0; s >>= 1) {
      if (tid < s && sv[tid + s] > sv[tid]) { sv[tid] = sv[tid + s]; si[tid] = si[tid + s]; }
      __syncthreads();
    }
    if (tid == 0) { chosen[pass] = si[0]; per[b * TB_NK + pass] = si[0]; }
    __syncthreads();
  }
}

// ---------------- 8a) depthwise 3/5/7 convs on folded image + GELU ---------
__global__ __launch_bounds__(256) void tb_dwconv(const float* __restrict__ hf,
                                                 const float* __restrict__ dw3,
                                                 const float* __restrict__ dw5,
                                                 const float* __restrict__ dw7,
                                                 const int* __restrict__ periods,
                                                 int b, int kidx,
                                                 _Float16* __restrict__ ybr) {
  const int gid = blockIdx.x * 256 + threadIdx.x;
  const int c  = gid & (TB_C - 1);
  const int nn = gid >> 7;                      // flat pixel index l = r*p + q
  int p = periods[b * TB_NK + kidx];
  if (p < 2) p = 2;
  const int Np  = (TB_L + p - 1) / p;
  const int Npp = Np * p;
  const float* hb = hf + (size_t)b * TB_L * TB_C;
  float a3 = 0.f, a5 = 0.f, a7 = 0.f;
  if (nn < Npp) {
    const int q = nn % p;
#pragma unroll
    for (int dr = -3; dr <= 3; ++dr) {
#pragma unroll
      for (int dq = -3; dq <= 3; ++dq) {
        const int qq = q + dq;
        const int l  = nn + dr * p + dq;
        if (qq < 0 || qq >= p || l < 0 || l >= Npp) continue;
        const float v = (l < TB_L) ? hb[(size_t)l * TB_C + c] : 0.f;
        a7 += v * dw7[c * 49 + (dr + 3) * 7 + (dq + 3)];
        if (dr >= -2 && dr <= 2 && dq >= -2 && dq <= 2)
          a5 += v * dw5[c * 25 + (dr + 2) * 5 + (dq + 2)];
        if (dr >= -1 && dr <= 1 && dq >= -1 && dq <= 1)
          a3 += v * dw3[c * 9 + (dr + 1) * 3 + (dq + 1)];
      }
    }
  }
  const size_t o = (size_t)nn * TB_C + c;
  ybr[o]                              = (_Float16)tb_gelu(a3);
  ybr[(size_t)TB_LMAX * TB_C + o]     = (_Float16)tb_gelu(a5);
  ybr[(size_t)2 * TB_LMAX * TB_C + o] = (_Float16)tb_gelu(a7);
}

// ---------------- 8b) pointwise 128->256 per branch, GELU, into z ----------
__global__ __launch_bounds__(256) void tb_pw(const _Float16* __restrict__ ybr,
                                             const _Float16* __restrict__ pwh, // 3 x (256,128) NK
                                             _Float16* __restrict__ z) {
  __shared__ __attribute__((aligned(16))) _Float16 sA[16 * TB_C];
  const int mt = blockIdx.x;
  const int br = blockIdx.z;
  const int wv = threadIdx.x >> 5;
  const int lane = threadIdx.x & 31;
  const int nt = blockIdx.y * 8 + wv;
  const _Float16* pw = pwh + (size_t)br * TB_HID * TB_C;
  tb_stage(sA, ybr + (size_t)br * TB_LMAX * TB_C + (size_t)mt * 16 * TB_C, 16 * TB_C);
  v8f c = {};
#pragma unroll
  for (int ks = 0; ks < 4; ++ks) {
    v16h a  = tb_frag(sA + ks * 32, TB_C, lane);
    v16h bf = tb_frag(pw + (size_t)(nt * 16) * TB_C + ks * 32, TB_C, lane);
    c = TB_WMMA(a, bf, c);
  }
  const int n  = lane & 15;
  const int rb = (lane >> 4) << 3;
#pragma unroll
  for (int r = 0; r < 8; ++r) {
    const int row = mt * 16 + rb + r;
    z[(size_t)row * (3 * TB_HID) + br * TB_HID + nt * 16 + n] = (_Float16)tb_gelu(c[r]);
  }
}

// ---------------- 8c) proj 768->128, unfold, accumulate into agg -----------
__global__ __launch_bounds__(256) void tb_proj(const _Float16* __restrict__ z,
                                               const _Float16* __restrict__ pjWh, // (128,768) NK
                                               const float* __restrict__ projB,
                                               int b,
                                               float* __restrict__ agg) {
  __shared__ __attribute__((aligned(16))) _Float16 sA[16 * 3 * TB_HID];  // 24 KB
  const int mt = blockIdx.x;
  const int wv = threadIdx.x >> 5;
  const int lane = threadIdx.x & 31;
  tb_stage(sA, z + (size_t)mt * 16 * (3 * TB_HID), 16 * 3 * TB_HID);
  v8f c = {};
  for (int ks = 0; ks < (3 * TB_HID) / 32; ++ks) {     // 24 k-steps
    v16h a  = tb_frag(sA + ks * 32, 3 * TB_HID, lane);
    v16h bf = tb_frag(pjWh + (size_t)(wv * 16) * (3 * TB_HID) + ks * 32, 3 * TB_HID, lane);
    c = TB_WMMA(a, bf, c);
  }
  const int n  = lane & 15;
  const int rb = (lane >> 4) << 3;
  const float bias = projB[wv * 16 + n];
#pragma unroll
  for (int r = 0; r < 8; ++r) {
    const int row = mt * 16 + rb + r;
    if (row < TB_L)                                    // unfold: keep first L rows
      agg[((size_t)b * TB_L + row) * TB_C + wv * 16 + n] += c[r] + bias;
  }
}

// ---------------- 9) aggh = (agg/3) as f16 ---------------------------------
__global__ __launch_bounds__(256) void tb_scale(const float* __restrict__ agg,
                                                _Float16* __restrict__ aggh) {
  const size_t i = (size_t)blockIdx.x * 256 + threadIdx.x;
  aggh[i] = (_Float16)(agg[i] * (1.0f / TB_NK));
}

// ---------------- 10) out = x + aggh @ pout_W + pout_b ---------------------
__global__ __launch_bounds__(256) void tb_final(const _Float16* __restrict__ aggh,
                                                const float* __restrict__ x,
                                                const _Float16* __restrict__ poWh, // (128,128) NK
                                                const float* __restrict__ poB,
                                                float* __restrict__ out) {
  __shared__ __attribute__((aligned(16))) _Float16 sA[16 * TB_C];
  const int mt = blockIdx.x;
  const int wv = threadIdx.x >> 5;
  const int lane = threadIdx.x & 31;
  tb_stage(sA, aggh + (size_t)mt * 16 * TB_C, 16 * TB_C);
  v8f c = {};
#pragma unroll
  for (int ks = 0; ks < 4; ++ks) {
    v16h a  = tb_frag(sA + ks * 32, TB_C, lane);
    v16h bf = tb_frag(poWh + (size_t)(wv * 16) * TB_C + ks * 32, TB_C, lane);
    c = TB_WMMA(a, bf, c);
  }
  const int n  = lane & 15;
  const int rb = (lane >> 4) << 3;
  const float bias = poB[wv * 16 + n];
#pragma unroll
  for (int r = 0; r < 8; ++r) {
    const size_t o = ((size_t)mt * 16 + rb + r) * TB_C + wv * 16 + n;
    out[o] = x[o] + c[r] + bias;
  }
}

extern "C" void kernel_launch(void* const* d_in, const int* in_sizes, int n_in,
                              void* d_out, int out_size, void* d_ws, size_t ws_size,
                              hipStream_t stream) {
  (void)in_sizes; (void)n_in; (void)out_size; (void)ws_size;
  const float* x      = (const float*)d_in[0];
  const float* norm_w = (const float*)d_in[1];
  const float* glu_W  = (const float*)d_in[2];
  const float* glu_b  = (const float*)d_in[3];
  const float* pin_W  = (const float*)d_in[4];
  const float* pin_b  = (const float*)d_in[5];
  const float* dw3    = (const float*)d_in[6];
  const float* pw3    = (const float*)d_in[7];
  const float* dw5    = (const float*)d_in[8];
  const float* pw5    = (const float*)d_in[9];
  const float* dw7    = (const float*)d_in[10];
  const float* pw7    = (const float*)d_in[11];
  const float* proj_W = (const float*)d_in[12];
  const float* proj_b = (const float*)d_in[13];
  const float* pout_W = (const float*)d_in[14];
  const float* pout_b = (const float*)d_in[15];
  float* out = (float*)d_out;

  char* w8 = (char*)d_ws;
  size_t off = 0;
  auto alloc = [&](size_t bytes) -> char* {
    char* p = w8 + off;
    off += (bytes + 255) & ~(size_t)255;
    return p;
  };
  _Float16* xn   = (_Float16*)alloc((size_t)TB_M * TB_C * sizeof(_Float16));     // 8 MB
  _Float16* h1   = (_Float16*)alloc((size_t)TB_M * TB_C * sizeof(_Float16));     // 8 MB
  float*    hf   = (float*)   alloc((size_t)TB_M * TB_C * sizeof(float));        // 16 MB
  float*    mn   = (float*)   alloc((size_t)TB_B * TB_C * sizeof(float));
  float*    ac   = (float*)   alloc((size_t)TB_B * TB_L * sizeof(float));
  int*      per  = (int*)     alloc(256);
  _Float16* x0   = (_Float16*)alloc((size_t)TB_M * TB_C * sizeof(_Float16));     // 8 MB (reused as aggh)
  _Float16* ybr  = (_Float16*)alloc((size_t)3 * TB_LMAX * TB_C * sizeof(_Float16));
  _Float16* z    = (_Float16*)alloc((size_t)TB_LMAX * 3 * TB_HID * sizeof(_Float16));
  float*    agg  = (float*)   alloc((size_t)TB_M * TB_C * sizeof(float));        // 16 MB
  _Float16* gwh  = (_Float16*)alloc((size_t)2 * TB_C * TB_C * sizeof(_Float16)); // (256,128)
  _Float16* pinh = (_Float16*)alloc((size_t)TB_C * TB_C * sizeof(_Float16));     // (128,128)
  _Float16* pouh = (_Float16*)alloc((size_t)TB_C * TB_C * sizeof(_Float16));     // (128,128)
  _Float16* pwh  = (_Float16*)alloc((size_t)3 * TB_HID * TB_C * sizeof(_Float16));
  _Float16* pjh  = (_Float16*)alloc((size_t)TB_C * 3 * TB_HID * sizeof(_Float16));

  // weight prep (tiny, one-shot per call)
  tb_prep_t<<<(2 * TB_C * TB_C) / 256, 256, 0, stream>>>(glu_W, gwh, TB_C, 2 * TB_C);
  tb_prep_t<<<(TB_C * TB_C) / 256, 256, 0, stream>>>(pin_W, pinh, TB_C, TB_C);
  tb_prep_t<<<(TB_C * TB_C) / 256, 256, 0, stream>>>(pout_W, pouh, TB_C, TB_C);
  tb_prep_c<<<(TB_HID * TB_C) / 256, 256, 0, stream>>>(pw3, pwh);
  tb_prep_c<<<(TB_HID * TB_C) / 256, 256, 0, stream>>>(pw5, pwh + (size_t)TB_HID * TB_C);
  tb_prep_c<<<(TB_HID * TB_C) / 256, 256, 0, stream>>>(pw7, pwh + (size_t)2 * TB_HID * TB_C);
  tb_prep_c<<<(TB_C * 3 * TB_HID) / 256, 256, 0, stream>>>(proj_W, pjh);

  tb_rms     <<<TB_M / 8, 256, 0, stream>>>(x, norm_w, xn);
  tb_glu     <<<TB_M / 16, 256, 0, stream>>>(xn, gwh, glu_b, h1);
  tb_pin     <<<TB_M / 16, 256, 0, stream>>>(h1, pinh, pin_b, hf);
  tb_mean    <<<(TB_B * TB_C) / 8, 256, 0, stream>>>(hf, mn);
  tb_x0agg   <<<(TB_M * TB_C) / 256, 256, 0, stream>>>(hf, mn, x0, agg);
  tb_zero_ac <<<(TB_B * TB_L) / 256, 256, 0, stream>>>(ac);
  tb_autocorr<<<TB_B * 256, 256, 0, stream>>>(x0, ac);
  tb_topk    <<<TB_B, 256, 0, stream>>>(ac, per);

  for (int b = 0; b < TB_B; ++b) {
    for (int k = 0; k < TB_NK; ++k) {
      tb_dwconv<<<(TB_LMAX * TB_C) / 256, 256, 0, stream>>>(hf, dw3, dw5, dw7, per, b, k, ybr);
      tb_pw    <<<dim3(TB_LMAX / 16, 2, 3), 256, 0, stream>>>(ybr, pwh, z);
      tb_proj  <<<TB_LMAX / 16, 256, 0, stream>>>(z, pjh, proj_b, b, agg);
    }
  }

  tb_scale <<<(TB_M * TB_C) / 256, 256, 0, stream>>>(agg, x0);
  tb_final <<<TB_M / 16, 256, 0, stream>>>(x0, x, pouh, pout_b, out);
}
